// GnLTransformer_Paired_19911468384991
// MI455X (gfx1250) — matrix operand order repaired and checked by reading, not compiled, not run on hardware
//
#include <hip/hip_runtime.h>
#include <hip/hip_bf16.h>
#include <math.h>

// ---------------- problem constants ----------------
constexpr int kN      = 32768;     // nodes
constexpr int kE      = 262144;    // edges
constexpr int kB      = 64;        // graphs
constexpr int kNPG    = 512;       // nodes per graph
constexpr int kK      = 20;        // top-k
constexpr int kDIN    = 64;
constexpr int kDH     = 128;
constexpr int kHC     = 256;       // heads*channels of conv1
constexpr int kHEADS  = 4;
constexpr int kEDIM   = 8;
constexpr int kDHL    = 256;
constexpr int kDOUT   = 8;

typedef __attribute__((ext_vector_type(16))) _Float16 v16h;
typedef __attribute__((ext_vector_type(8)))  float    v8f;

// ---- CDNA5 async LDS copy path (GLOBAL_LOAD_ASYNC_TO_LDS + s_wait_asynccnt) ----
#if defined(__has_builtin)
#if __has_builtin(__builtin_amdgcn_global_load_async_to_lds_b128) && \
    __has_builtin(__builtin_amdgcn_s_wait_asynccnt)
#define CDNA5_ASYNC_LDS 1
#endif
#endif
#ifndef CDNA5_ASYNC_LDS
#define CDNA5_ASYNC_LDS 0
#endif

typedef int i32x4 __attribute__((ext_vector_type(4)));
typedef __attribute__((address_space(1))) i32x4 as1_i32x4;   // global
typedef __attribute__((address_space(3))) i32x4 as3_i32x4;   // LDS

// ---------------- small helpers ----------------
__device__ __forceinline__ void atomicMaxF(float* addr, float val) {
  unsigned int* ua = (unsigned int*)addr;
  unsigned int old = *ua;
  for (;;) {
    float cur = __uint_as_float(old);
    if (cur >= val) break;
    unsigned int assumed = old;
    old = atomicCAS(ua, assumed, __float_as_uint(val));
    if (old == assumed) break;
  }
}

__device__ __forceinline__ float act_apply(float x, int act, float slope) {
  if (act == 1) return fmaxf(x, 0.0f);
  if (act == 2) return x > 0.0f ? x : slope * x;
  if (act == 3) return x > 0.0f ? x : (expf(x) - 1.0f);   // elu
  return x;
}

// ---------------- conversion / elementwise ----------------
__global__ void cvt_f32_f16(const float* __restrict__ s, _Float16* __restrict__ d, int n) {
  int i = blockIdx.x * blockDim.x + threadIdx.x;
  if (i < n) d[i] = (_Float16)s[i];
}

__global__ void cvt_block_f16(const float* __restrict__ s, _Float16* __restrict__ d,
                              int rows, int cols, int dstStride, int dstOff) {
  int i = blockIdx.x * blockDim.x + threadIdx.x;
  if (i >= rows * cols) return;
  int r = i / cols, c = i - r * cols;
  d[(size_t)r * dstStride + dstOff + c] = (_Float16)s[i];
}

__global__ void fill_kernel(float* __restrict__ p, float v, int n) {
  int i = blockIdx.x * blockDim.x + threadIdx.x;
  if (i < n) p[i] = v;
}

__global__ void add_inplace(float* __restrict__ dst, const float* __restrict__ src, int n) {
  int i = blockIdx.x * blockDim.x + threadIdx.x;
  if (i < n) dst[i] += src[i];
}

__global__ void bias_act_kernel(float* __restrict__ x, const float* __restrict__ b,
                                int rows, int cols, int act, float slope) {
  int i = blockIdx.x * blockDim.x + threadIdx.x;
  if (i >= rows * cols) return;
  int c = i % cols;
  x[i] = act_apply(x[i] + b[c], act, slope);
}

// ---------------- WMMA GEMM: C[M,Nc] = act( A[M,Kc] @ W[Nc,Kc]^T + bias (+C) ) ----------------
// 4 waves / block; each wave owns one 16x16 output tile (4 M-tiles per block, one N-tile).
// The 16-row weight panel for the N-tile is staged once into LDS with async copies and
// shared by all 4 waves; K stepped by 32 with v_wmma_f32_16x16x32_f16.

__device__ __forceinline__ void wmma_kloop(const _Float16* __restrict__ Arow,
                                           const _Float16* __restrict__ Wrow,
                                           int Kc, int akb, int bkb, v8f& acc) {
#pragma unroll 2
  for (int k0 = 0; k0 < Kc; k0 += 32) {
    __builtin_prefetch(Arow + k0 + 64, 0, 1);   // global_prefetch_b8
    v16h a, b;
#pragma unroll
    for (int t = 0; t < 16; ++t) {
      const int v = t >> 1, p = t & 1;
      const int ka = akb + 2 * v + p;
      a[t] = Arow[k0 + ka];
      b[t] = Wrow[k0 + bkb + 2 * v + p];
    }
    acc = __builtin_amdgcn_wmma_f32_16x16x32_f16(false, a, false, b, (short)0, acc, false, false);
  }
}

__global__ void wmma_gemm(const _Float16* __restrict__ A, const _Float16* __restrict__ W,
                          const float* __restrict__ bias, float* __restrict__ C,
                          int M, int Nc, int Kc, int act, float slope, int accum) {
#if CDNA5_ASYNC_LDS
  __shared__ alignas(16) _Float16 sW[16 * 1024];   // 32KB: weight panel [16][Kc<=1024]
#endif
  const int lane = threadIdx.x & 31;
  const int wv   = threadIdx.x >> 5;
  const int i0   = blockIdx.x * 64 + wv * 16;   // M tile of this wave
  const int j0   = blockIdx.y * 16;             // N tile of this block
  const int nloc = lane & 15;
  const int hi   = (lane >> 4) & 1;
  const int n    = j0 + nloc;

  bool lds_ok = false;
#if CDNA5_ASYNC_LDS
  if (Kc <= 1024) {
    // 16 contiguous rows of W -> one contiguous 16*Kc*2 byte region; B128 async copy.
    const char* g = (const char*)(W + (size_t)j0 * Kc);
    char*       l = (char*)&sW[0];
    const int nbytes = 16 * Kc * 2;
    for (int o = (int)threadIdx.x * 16; o < nbytes; o += (int)blockDim.x * 16) {
      __builtin_amdgcn_global_load_async_to_lds_b128(
          (as1_i32x4*)(const void*)(g + o), (as3_i32x4*)(void*)(l + o), 0, 0);
    }
    __builtin_amdgcn_s_wait_asynccnt(0);
    __syncthreads();
    lds_ok = true;
  }
#endif

  v8f acc;
  const float binit = bias ? bias[n] : 0.0f;
#pragma unroll
  for (int r = 0; r < 8; ++r) {
    const int m = i0 + r + hi * 8;
    acc[r] = binit + (accum ? C[(size_t)m * Nc + n] : 0.0f);
  }

  const int arow = i0 + nloc;
  const int akb  = hi ? 16 : 0;            // A: lanes 16-31 hold K=16..31
  const int bkb  = hi ? 16 : 0;            // B: lanes 16-31 hold K=16..31
  const _Float16* __restrict__ Arow = A + (size_t)arow * Kc;

  if (lds_ok) {
#if CDNA5_ASYNC_LDS
    const _Float16* Wl = &sW[(size_t)nloc * Kc];   // ds_load path
    wmma_kloop(Arow, Wl, Kc, akb, bkb, acc);
#endif
  } else {
    const _Float16* __restrict__ Wrow = W + (size_t)n * Kc;
    wmma_kloop(Arow, Wrow, Kc, akb, bkb, acc);
  }

#pragma unroll
  for (int r = 0; r < 8; ++r) {
    const int m = i0 + r + hi * 8;
    C[(size_t)m * Nc + n] = act_apply(acc[r], act, slope);
  }
}

// ---------------- TransformerConv edge kernels (edge lin fused, K=8) ----------------
__global__ void tc_alpha(const float* __restrict__ q, const float* __restrict__ k,
                         const float* __restrict__ ea, const float* __restrict__ We,
                         const int* __restrict__ src, const int* __restrict__ dst,
                         float* __restrict__ alpha, int E_) {
  int tid = blockIdx.x * blockDim.x + threadIdx.x;
  if (tid >= E_ * kHEADS) return;
  const int e = tid >> 2, h = tid & 3;
  const int s = src[e], d = dst[e];
  float ea8[kEDIM];
#pragma unroll
  for (int j = 0; j < kEDIM; ++j) ea8[j] = ea[(size_t)e * kEDIM + j];
  const float* __restrict__ qr = q + (size_t)d * kHC + h * 64;
  const float* __restrict__ kr = k + (size_t)s * kHC + h * 64;
  const float* __restrict__ Wr = We + (size_t)(h * 64) * kEDIM;
  float acc = 0.0f;
  for (int c = 0; c < 64; ++c) {
    float ee = 0.0f;
#pragma unroll
    for (int j = 0; j < kEDIM; ++j) ee += ea8[j] * Wr[c * kEDIM + j];
    acc += qr[c] * (kr[c] + ee);
  }
  alpha[tid] = acc * 0.125f;   // / sqrt(64)
}

__global__ void tc_message(const float* __restrict__ v, const float* __restrict__ ea,
                           const float* __restrict__ We, const int* __restrict__ src,
                           const int* __restrict__ dst, const float* __restrict__ alpha,
                           float* __restrict__ out, int E_) {
  int tid = blockIdx.x * blockDim.x + threadIdx.x;
  if (tid >= E_ * kHEADS) return;
  const int e = tid >> 2, h = tid & 3;
  const int s = src[e], d = dst[e];
  const float al = alpha[tid];
  float ea8[kEDIM];
#pragma unroll
  for (int j = 0; j < kEDIM; ++j) ea8[j] = ea[(size_t)e * kEDIM + j];
  const float* __restrict__ vr = v + (size_t)s * kHC + h * 64;
  const float* __restrict__ Wr = We + (size_t)(h * 64) * kEDIM;
  float* __restrict__ orow = out + (size_t)d * kHC + h * 64;
  for (int c = 0; c < 64; ++c) {
    float ee = 0.0f;
#pragma unroll
    for (int j = 0; j < kEDIM; ++j) ee += ea8[j] * Wr[c * kEDIM + j];
    atomicAdd(&orow[c], (vr[c] + ee) * al);
  }
}

// ---------------- GATv2 edge kernels ----------------
__global__ void gat_alpha(const float* __restrict__ xl, const float* __restrict__ xr,
                          const float* __restrict__ ea, const float* __restrict__ We,
                          const float* __restrict__ att, const int* __restrict__ src,
                          const int* __restrict__ dst, float* __restrict__ alpha, int E_) {
  int e = blockIdx.x * blockDim.x + threadIdx.x;
  if (e >= E_) return;
  const int s = src[e], d = dst[e];
  float ea8[kEDIM];
#pragma unroll
  for (int j = 0; j < kEDIM; ++j) ea8[j] = ea[(size_t)e * kEDIM + j];
  const float* __restrict__ lr = xl + (size_t)s * kDH;
  const float* __restrict__ rr = xr + (size_t)d * kDH;
  float acc = 0.0f;
  for (int c = 0; c < kDH; ++c) {
    float ee = 0.0f;
#pragma unroll
    for (int j = 0; j < kEDIM; ++j) ee += ea8[j] * We[c * kEDIM + j];
    float z = lr[c] + rr[c] + ee;
    z = z > 0.0f ? z : 0.2f * z;
    acc += z * att[c];
  }
  alpha[e] = acc;
}

__global__ void gat_message(const float* __restrict__ xl, const float* __restrict__ alpha,
                            const int* __restrict__ src, const int* __restrict__ dst,
                            float* __restrict__ out, int E_) {
  int e = blockIdx.x * blockDim.x + threadIdx.x;
  if (e >= E_) return;
  const int s = src[e], d = dst[e];
  const float al = alpha[e];
  const float* __restrict__ lr = xl + (size_t)s * kDH;
  float* __restrict__ orow = out + (size_t)d * kDH;
  for (int c = 0; c < kDH; ++c) atomicAdd(&orow[c], lr[c] * al);
}

// ---------------- segment softmax (over dst), H heads ----------------
__global__ void seg_max_k(const float* __restrict__ a, const int* __restrict__ dst,
                          float* __restrict__ nmax, int E_, int H) {
  int i = blockIdx.x * blockDim.x + threadIdx.x;
  if (i >= E_ * H) return;
  int e = i / H, h = i - e * H;
  atomicMaxF(&nmax[(size_t)dst[e] * H + h], a[i]);
}
__global__ void seg_expsum_k(float* __restrict__ a, const int* __restrict__ dst,
                             const float* __restrict__ nmax, float* __restrict__ nsum,
                             int E_, int H) {
  int i = blockIdx.x * blockDim.x + threadIdx.x;
  if (i >= E_ * H) return;
  int e = i / H, h = i - e * H;
  float v = expf(a[i] - nmax[(size_t)dst[e] * H + h]);
  a[i] = v;
  atomicAdd(&nsum[(size_t)dst[e] * H + h], v);
}
__global__ void seg_norm_k(float* __restrict__ a, const int* __restrict__ dst,
                           const float* __restrict__ nsum, int E_, int H) {
  int i = blockIdx.x * blockDim.x + threadIdx.x;
  if (i >= E_ * H) return;
  int e = i / H, h = i - e * H;
  a[i] = a[i] / (nsum[(size_t)dst[e] * H + h] + 1e-16f);
}

// ---------------- GRU combine (elementwise) ----------------
__global__ void gru_combine(const float* __restrict__ gi, const float* __restrict__ gh,
                            const float* __restrict__ hprev, float* __restrict__ out,
                            int n, int H, int relu) {
  int i = blockIdx.x * blockDim.x + threadIdx.x;
  if (i >= n * H) return;
  int node = i / H, c = i - node * H;
  size_t b3 = (size_t)node * 3 * H;
  float ir = gi[b3 + c], iz = gi[b3 + H + c], in = gi[b3 + 2 * H + c];
  float hr = gh[b3 + c], hz = gh[b3 + H + c], hn = gh[b3 + 2 * H + c];
  float r = 1.0f / (1.0f + expf(-(ir + hr)));
  float z = 1.0f / (1.0f + expf(-(iz + hz)));
  float ng = tanhf(in + r * hn);
  float o = (1.0f - z) * ng + z * hprev[i];
  out[i] = relu ? fmaxf(o, 0.0f) : o;
}

// ---------------- SAG pool ----------------
__global__ void pool_pre(const float* __restrict__ x, const float* __restrict__ Wrel,
                         const float* __restrict__ Wroot, float* __restrict__ t,
                         float* __restrict__ r0, int n) {
  int i = blockIdx.x * blockDim.x + threadIdx.x;
  if (i >= n) return;
  const float* xr = x + (size_t)i * kDH;
  float a = 0.0f, b = 0.0f;
  for (int c = 0; c < kDH; ++c) { a += xr[c] * Wrel[c]; b += xr[c] * Wroot[c]; }
  t[i] = a; r0[i] = b;
}
__global__ void pool_edge(const float* __restrict__ t, const int* __restrict__ src,
                          const int* __restrict__ dst, float* __restrict__ acc, int E_) {
  int e = blockIdx.x * blockDim.x + threadIdx.x;
  if (e >= E_) return;
  atomicAdd(&acc[dst[e]], t[src[e]]);
}
__global__ void pool_score(const float* __restrict__ acc, const float* __restrict__ r0,
                           const float* __restrict__ brel, float* __restrict__ score, int n) {
  int i = blockIdx.x * blockDim.x + threadIdx.x;
  if (i >= n) return;
  score[i] = acc[i] + brel[0] + r0[i];
}

__global__ void topk_kernel(const float* __restrict__ score, float* __restrict__ topv,
                            int* __restrict__ topi) {
  __shared__ float sv[512];
  __shared__ float rv[256];
  __shared__ int   ri[256];
  int b = blockIdx.x, t = threadIdx.x;    // 256 threads per graph
  sv[t]       = score[b * kNPG + t];
  sv[t + 256] = score[b * kNPG + 256 + t];
  __syncthreads();
  for (int it = 0; it < kK; ++it) {
    float v0 = sv[t]; int i0 = t;
    float v1 = sv[t + 256];
    if (v1 > v0) { v0 = v1; i0 = t + 256; }
    rv[t] = v0; ri[t] = i0;
    __syncthreads();
    for (int s = 128; s > 0; s >>= 1) {
      if (t < s) {
        float a = rv[t], bb = rv[t + s];
        int   ia = ri[t], ib = ri[t + s];
        if (bb > a || (bb == a && ib < ia)) { rv[t] = bb; ri[t] = ib; }
      }
      __syncthreads();
    }
    if (t == 0) {
      topv[b * kK + it] = rv[0];
      topi[b * kK + it] = ri[0];
      sv[ri[0]] = -3.0e38f;
    }
    __syncthreads();
  }
}

__global__ void gather_topk(const float* __restrict__ x, const float* __restrict__ topv,
                            const int* __restrict__ topi, float* __restrict__ xk) {
  int i = blockIdx.x * blockDim.x + threadIdx.x;
  if (i >= kB * kK * kDH) return;
  int b = i / (kK * kDH);
  int r = (i / kDH) % kK;
  int c = i % kDH;
  int node = b * kNPG + topi[b * kK + r];
  xk[i] = x[(size_t)node * kDH + c] * tanhf(topv[b * kK + r]);
}

__global__ void sort_agg_kernel(const float* __restrict__ xk, float* __restrict__ pooled) {
  __shared__ float key[kK];
  __shared__ int   ord[kK];
  int b = blockIdx.x, t = threadIdx.x;    // 128 threads
  if (t < kK) key[t] = xk[((size_t)b * kK + t) * kDH + (kDH - 1)];
  __syncthreads();
  if (t == 0) {
    bool used[kK];
    for (int i = 0; i < kK; ++i) used[i] = false;
    for (int i = 0; i < kK; ++i) {
      int best = -1; float bv = 0.0f;
      for (int j = 0; j < kK; ++j)
        if (!used[j] && (best < 0 || key[j] > bv)) { bv = key[j]; best = j; }
      used[best] = true; ord[i] = best;
    }
  }
  __syncthreads();
  for (int i = 0; i < kK; ++i)
    pooled[(size_t)b * (kK * kDH) + i * kDH + t] = xk[((size_t)b * kK + ord[i]) * kDH + t];
}

// ---------------- MLP tail ----------------
__global__ void bn_relu(float* __restrict__ x, const float* __restrict__ g,
                        const float* __restrict__ b, int rows, int cols) {
  int j = blockIdx.x * blockDim.x + threadIdx.x;
  if (j >= cols) return;
  float mu = 0.0f;
  for (int r = 0; r < rows; ++r) mu += x[(size_t)r * cols + j];
  mu /= rows;
  float var = 0.0f;
  for (int r = 0; r < rows; ++r) { float d = x[(size_t)r * cols + j] - mu; var += d * d; }
  var /= rows;
  float inv = rsqrtf(var + 1e-5f);
  for (int r = 0; r < rows; ++r) {
    float v = (x[(size_t)r * cols + j] - mu) * inv * g[j] + b[j];
    x[(size_t)r * cols + j] = fmaxf(v, 0.0f);
  }
}

__global__ void final_linear(const float* __restrict__ x, const float* __restrict__ W,
                             const float* __restrict__ b, float* __restrict__ out,
                             int rows, int Kc, int Oc) {
  int i = blockIdx.x * blockDim.x + threadIdx.x;
  if (i >= rows * Oc) return;
  int r = i / Oc, o = i - r * Oc;
  float acc = b[o];
  for (int k = 0; k < Kc; ++k) acc += x[(size_t)r * Kc + k] * W[(size_t)o * Kc + k];
  out[i] = acc;
}

// ---------------- host-side plumbing ----------------
namespace {

struct Bufs {
  float *fA, *fB, *fC, *fD, *fG, *eAl, *nMax, *nSum;
  _Float16 *a16_0, *a16_1, *w16, *m16;
  float *pT, *pR, *pAcc, *pScore, *topv, *xk, *pooledG, *pooledL, *z1, *z2;
  int *topi;
};

inline int cdiv(int a, int b) { return (a + b - 1) / b; }

inline void cvt(hipStream_t st, const float* s, _Float16* d, int n) {
  cvt_f32_f16<<<cdiv(n, 256), 256, 0, st>>>(s, d, n);
}

inline void gemm(hipStream_t st, Bufs& bf, const _Float16* A, const float* Wf,
                 const float* bias, float* C, int M, int Nc, int Kc,
                 int act, float slope, int accum) {
  cvt(st, Wf, bf.w16, Nc * Kc);
  wmma_gemm<<<dim3(M / 64, Nc / 16), dim3(128), 0, st>>>(A, bf.w16, bias, C,
                                                         M, Nc, Kc, act, slope, accum);
}

inline void seg_softmax(hipStream_t st, Bufs& bf, float* alpha, const int* dst, int H) {
  const int NH = kN * H, EH = kE * H;
  fill_kernel<<<cdiv(NH, 256), 256, 0, st>>>(bf.nMax, -3.0e38f, NH);
  (void)hipMemsetAsync(bf.nSum, 0, (size_t)NH * 4, st);
  seg_max_k<<<cdiv(EH, 256), 256, 0, st>>>(alpha, dst, bf.nMax, kE, H);
  seg_expsum_k<<<cdiv(EH, 256), 256, 0, st>>>(alpha, dst, bf.nMax, bf.nSum, kE, H);
  seg_norm_k<<<cdiv(EH, 256), 256, 0, st>>>(alpha, dst, bf.nSum, kE, H);
}

// One branch of attentive_conv; result (g) left in bf.fG.
inline void run_branch(hipStream_t st, Bufs& bf, void* const* d_in, int pb,
                       const float* x, const int* src, const int* dst, const float* ea) {
  auto P = [&](int i) { return (const float*)d_in[pb + i]; };
  const int NE4 = kE * kHEADS;

  // x -> f16  [N, 64]
  cvt(st, x, bf.a16_0, kN * kDIN);

  float* q = bf.fA;
  float* k = bf.fA + (size_t)kN * kHC;
  float* v = bf.fA + 2 * (size_t)kN * kHC;
  gemm(st, bf, bf.a16_0, P(0), P(1), q, kN, kHC, kDIN, 0, 0.f, 0);   // Wq,bq
  gemm(st, bf, bf.a16_0, P(2), P(3), k, kN, kHC, kDIN, 0, 0.f, 0);   // Wk,bk
  gemm(st, bf, bf.a16_0, P(4), P(5), v, kN, kHC, kDIN, 0, 0.f, 0);   // Wv,bv

  tc_alpha<<<cdiv(NE4, 256), 256, 0, st>>>(q, k, ea, P(6), src, dst, bf.eAl, kE);
  seg_softmax(st, bf, bf.eAl, dst, kHEADS);

  (void)hipMemsetAsync(bf.fC, 0, (size_t)kN * kHC * 4, st);
  tc_message<<<cdiv(NE4, 256), 256, 0, st>>>(v, ea, P(6), src, dst, bf.eAl, bf.fC, kE);
  // skip connection + fused elu:  h = elu(msg + x@Wskip^T + bskip)
  gemm(st, bf, bf.a16_0, P(7), P(8), bf.fC, kN, kHC, kDIN, 3, 0.f, 1);

  // lin0: x0 = leaky_relu(x@W^T+b, 0.01)
  gemm(st, bf, bf.a16_0, P(9), P(10), bf.fD, kN, kHC, kDIN, 2, 0.01f, 0);

  // gru1: gi = h@Wih^T+bih ; gh = x0@Whh^T+bhh ; x = relu(gru(...))
  cvt(st, bf.fC, bf.a16_1, kN * kHC);
  cvt(st, bf.fD, bf.a16_0, kN * kHC);
  gemm(st, bf, bf.a16_1, P(11), P(13), bf.fA, kN, 3 * kHC, kHC, 0, 0.f, 0);
  gemm(st, bf, bf.a16_0, P(12), P(14), bf.fB, kN, 3 * kHC, kHC, 0, 0.f, 0);
  gru_combine<<<cdiv(kN * kHC, 256), 256, 0, st>>>(bf.fA, bf.fB, bf.fD, bf.fD, kN, kHC, 1);

  // lin1: x = leaky_relu(x@W^T+b, 0.01)   [N,128] -> fC
  cvt(st, bf.fD, bf.a16_0, kN * kHC);
  gemm(st, bf, bf.a16_0, P(15), P(16), bf.fC, kN, kDH, kHC, 2, 0.01f, 0);

  // g = x
  (void)hipMemcpyAsync(bf.fG, bf.fC, (size_t)kN * kDH * 4, hipMemcpyDeviceToDevice, st);

  for (int t = 0; t < 2; ++t) {
    const int cb = 17 + t * 7;   // Wl,bl,Wr,br,We,att,bias
    const int gb = 31 + t * 4;   // Wih,Whh,bih,bhh
    cvt(st, bf.fC, bf.a16_0, kN * kDH);

    float* xl = bf.fA;
    float* xr = bf.fA + (size_t)kN * kDH;
    float* gi = bf.fA + 2 * (size_t)kN * kDH;      // N*384 floats
    float* hG = bf.fB;
    float* gh = bf.fB + (size_t)kN * kDH;          // N*384 floats
    gemm(st, bf, bf.a16_0, P(cb + 0), P(cb + 1), xl, kN, kDH, kDH, 0, 0.f, 0);
    gemm(st, bf, bf.a16_0, P(cb + 2), P(cb + 3), xr, kN, kDH, kDH, 0, 0.f, 0);

    gat_alpha<<<cdiv(kE, 256), 256, 0, st>>>(xl, xr, ea, P(cb + 4), P(cb + 5),
                                             src, dst, bf.eAl, kE);
    seg_softmax(st, bf, bf.eAl, dst, 1);

    (void)hipMemsetAsync(hG, 0, (size_t)kN * kDH * 4, st);
    gat_message<<<cdiv(kE, 256), 256, 0, st>>>(xl, bf.eAl, src, dst, hG, kE);
    bias_act_kernel<<<cdiv(kN * kDH, 256), 256, 0, st>>>(hG, P(cb + 6), kN, kDH, 3, 0.f);

    cvt(st, hG, bf.a16_1, kN * kDH);
    gemm(st, bf, bf.a16_1, P(gb + 0), P(gb + 2), gi, kN, 3 * kDH, kDH, 0, 0.f, 0);
    gemm(st, bf, bf.a16_0, P(gb + 1), P(gb + 3), gh, kN, 3 * kDH, kDH, 0, 0.f, 0);
    gru_combine<<<cdiv(kN * kDH, 256), 256, 0, st>>>(gi, gh, bf.fC, bf.fC, kN, kDH, 1);

    add_inplace<<<cdiv(kN * kDH, 256), 256, 0, st>>>(bf.fG, bf.fC, kN * kDH);
  }
}

// SAGPool + sort-aggregation; reads bf.fG, writes pooled [B, K*DH]
inline void run_pool(hipStream_t st, Bufs& bf, void* const* d_in, int pb,
                     const int* src, const int* dst, float* pooled) {
  auto P = [&](int i) { return (const float*)d_in[pb + i]; };
  pool_pre<<<cdiv(kN, 256), 256, 0, st>>>(bf.fG, P(0), P(2), bf.pT, bf.pR, kN);
  (void)hipMemsetAsync(bf.pAcc, 0, (size_t)kN * 4, st);
  pool_edge<<<cdiv(kE, 256), 256, 0, st>>>(bf.pT, src, dst, bf.pAcc, kE);
  pool_score<<<cdiv(kN, 256), 256, 0, st>>>(bf.pAcc, bf.pR, P(1), bf.pScore, kN);
  topk_kernel<<<kB, 256, 0, st>>>(bf.pScore, bf.topv, bf.topi);
  gather_topk<<<cdiv(kB * kK * kDH, 256), 256, 0, st>>>(bf.fG, bf.topv, bf.topi, bf.xk);
  sort_agg_kernel<<<kB, kDH, 0, st>>>(bf.xk, pooled);
}

} // namespace

extern "C" void kernel_launch(void* const* d_in, const int* in_sizes, int n_in,
                              void* d_out, int out_size, void* d_ws, size_t ws_size,
                              hipStream_t stream) {
  (void)in_sizes; (void)n_in; (void)out_size; (void)ws_size;

  // ---- inputs (setup_inputs order) ----
  const float* x_G  = (const float*)d_in[0];
  const int*   ei_G = (const int*)  d_in[1];   // [2,E] flat: src then dst
  const float* ea_G = (const float*)d_in[2];
  const float* x_L  = (const float*)d_in[3];
  const int*   ei_L = (const int*)  d_in[4];
  const float* ea_L = (const float*)d_in[5];
  // param leaves flattened depth-first in dict insertion order:
  const int PB_G = 6, PB_L = 45, PB_PG = 84, PB_PL = 87, PB_MLP = 90;

  // ---- workspace layout ----
  char* ws = (char*)d_ws;
  size_t off = 0;
  auto take = [&](size_t bytes) { size_t r = off; off += (bytes + 255) & ~(size_t)255; return r; };
  Bufs bf;
  bf.fA      = (float*)(ws + take((size_t)kN * 768 * 4));
  bf.fB      = (float*)(ws + take((size_t)kN * 768 * 4));
  bf.fC      = (float*)(ws + take((size_t)kN * kHC * 4));
  bf.fD      = (float*)(ws + take((size_t)kN * kHC * 4));
  bf.fG      = (float*)(ws + take((size_t)kN * kDH * 4));
  bf.eAl     = (float*)(ws + take((size_t)kE * kHEADS * 4));
  bf.nMax    = (float*)(ws + take((size_t)kN * kHEADS * 4));
  bf.nSum    = (float*)(ws + take((size_t)kN * kHEADS * 4));
  bf.a16_0   = (_Float16*)(ws + take((size_t)kN * kHC * 2));
  bf.a16_1   = (_Float16*)(ws + take((size_t)kN * kHC * 2));
  bf.w16     = (_Float16*)(ws + take((size_t)4 << 20));     // fits 256x5120 f16
  bf.m16     = (_Float16*)(ws + take((size_t)kB * 5120 * 2));
  bf.pT      = (float*)(ws + take((size_t)kN * 4));
  bf.pR      = (float*)(ws + take((size_t)kN * 4));
  bf.pAcc    = (float*)(ws + take((size_t)kN * 4));
  bf.pScore  = (float*)(ws + take((size_t)kN * 4));
  bf.topv    = (float*)(ws + take((size_t)kB * kK * 4));
  bf.topi    = (int*)  (ws + take((size_t)kB * kK * 4));
  bf.xk      = (float*)(ws + take((size_t)kB * kK * kDH * 4));
  bf.pooledG = (float*)(ws + take((size_t)kB * kK * kDH * 4));
  bf.pooledL = (float*)(ws + take((size_t)kB * kK * kDH * 4));
  bf.z1      = (float*)(ws + take((size_t)kB * kDHL * 4));
  bf.z2      = (float*)(ws + take((size_t)kB * kDHL * 4));

  const int* srcG = ei_G;            const int* dstG = ei_G + kE;
  const int* srcL = ei_L;            const int* dstL = ei_L + kE;

  // ---- branch G -> pool G ----
  run_branch(stream, bf, d_in, PB_G, x_G, srcG, dstG, ea_G);
  run_pool(stream, bf, d_in, PB_PG, srcG, dstG, bf.pooledG);

  // ---- branch L -> pool L (buffers reused) ----
  run_branch(stream, bf, d_in, PB_L, x_L, srcL, dstL, ea_L);
  run_pool(stream, bf, d_in, PB_PL, srcL, dstL, bf.pooledL);

  // ---- MLP head ----
  auto PM = [&](int i) { return (const float*)d_in[PB_MLP + i]; };
  const int CIN = 2 * kK * kDH;  // 5120
  cvt_block_f16<<<cdiv(kB * kK * kDH, 256), 256, 0, stream>>>(bf.pooledG, bf.m16,
                                                              kB, kK * kDH, CIN, 0);
  cvt_block_f16<<<cdiv(kB * kK * kDH, 256), 256, 0, stream>>>(bf.pooledL, bf.m16,
                                                              kB, kK * kDH, CIN, kK * kDH);
  gemm(stream, bf, bf.m16, PM(0), PM(1), bf.z1, kB, kDHL, CIN, 0, 0.f, 0);   // lin0
  bn_relu<<<cdiv(kDHL, 256), 256, 0, stream>>>(bf.z1, PM(6), PM(7), kB, kDHL);
  cvt(stream, bf.z1, bf.m16, kB * kDHL);
  gemm(stream, bf, bf.m16, PM(2), PM(3), bf.z2, kB, kDHL, kDHL, 0, 0.f, 0);  // lin1
  bn_relu<<<cdiv(kDHL, 256), 256, 0, stream>>>(bf.z2, PM(8), PM(9), kB, kDHL);
  final_linear<<<cdiv(kB * kDOUT, 256), 256, 0, stream>>>(bf.z2, PM(4), PM(5),
                                                          (float*)d_out, kB, kDHL, kDOUT);
}
